// TransformerBlock_29695403885288
// MI455X (gfx1250) — matrix operand order repaired and checked by reading, not compiled, not run on hardware
//
#include <hip/hip_runtime.h>
#include <hip/hip_bf16.h>
#include <math.h>

// ---------------------------------------------------------------------------
// Transformer block for MI455X (gfx1250, wave32, WMMA).
// Split-bf16 (hi/lo limb) WMMA for near-fp32 accuracy. All fp32->bf16
// conversion + fragment swizzling happens exactly once per element in
// producer epilogues / a one-shot weight-convert kernel. The GEMM k-loop is
// ping-pong double-buffered with a deliberately small per-thread staging
// footprint (3 float4 regs) so nothing spills to scratch.
// ---------------------------------------------------------------------------

#define Dm   1024
#define Hh   16
#define HDim 64
#define DFFn 4096
#define Bb   2
#define Ss   2048
#define Mrows (Bb * Ss)

typedef __attribute__((ext_vector_type(16))) __bf16 v16bf;
typedef __attribute__((ext_vector_type(8)))  float  v8f;

__device__ __forceinline__ void split_bf16(float x, __bf16& hi, __bf16& lo) {
  hi = (__bf16)x;
  lo = (__bf16)(x - (float)hi);
}

// ---------------------------------------------------------------------------
// Fragment-order ("swizzled") layouts. A fragment is 16 contiguous bf16 per
// lane; a 16(row/col)x32(k) strip is 32 lanes * 16 elems = 512 bf16.
//
// A operand (16x32):             lane = (row&15) + 16*((k&31)>>4),
//                                e    = k & 15
// B operand (32x16):             lane = (n&15) + 16*((k&31)>>4),
//                                e    = k & 15
// ---------------------------------------------------------------------------
__device__ __forceinline__ size_t a_sw(int row, int k, int Kd) {
  const int kk = k & 31;
  const size_t strip = (size_t)(row >> 4) * (Kd >> 5) + (k >> 5);
  const int lane = (row & 15) + ((kk >> 4) << 4);
  const int e = kk & 15;
  return strip * 512 + lane * 16 + e;
}
__device__ __forceinline__ size_t b_sw(int k, int n, int Kd) {
  const int kk = k & 31;
  const size_t strip = (size_t)(n >> 4) * (Kd >> 5) + (k >> 5);
  const int lane = (n & 15) + ((kk >> 4) << 4);
  const int e = kk & 15;
  return strip * 512 + lane * 16 + e;
}
// K operand for Q@K^T (B operand, n = seq row, k = hd within head).
__device__ __forceinline__ size_t k_sw(int row, int col) {
  const int h = col >> 6, hd = col & 63;
  const int kk = hd & 31;
  const size_t strip = (size_t)(row >> 4) * (Dm >> 5) + h * 2 + (hd >> 5);
  const int lane = (row & 15) + ((kk >> 4) << 4);
  const int e = kk & 15;
  return strip * 512 + lane * 16 + e;
}
// V operand for P@V (B operand, n = hd, k = seq pos).
__device__ __forceinline__ size_t v_sw(int row, int col) {
  const int b = row >> 11, s = row & (Ss - 1);
  const int h = col >> 6, hd = col & 63;
  const size_t strip = ((size_t)(b * Hh + h) * 4 + (hd >> 4)) * (Ss >> 5) + (s >> 5);
  const int lane = (hd & 15) + (((s & 31) >> 4) << 4);
  const int e = s & 15;
  return strip * 512 + lane * 16 + e;
}

// Split-precision product: Ah*Bh + Ah*Bl + Al*Bh (drops only lo*lo ~ 2^-16).
__device__ __forceinline__ v8f wmma3(v16bf ah, v16bf al, v16bf bh, v16bf bl, v8f c) {
  c = __builtin_amdgcn_wmma_f32_16x16x32_bf16(false, ah, false, bh, (short)0, c, false, false);
  c = __builtin_amdgcn_wmma_f32_16x16x32_bf16(false, ah, false, bl, (short)0, c, false, false);
  c = __builtin_amdgcn_wmma_f32_16x16x32_bf16(false, al, false, bh, (short)0, c, false, false);
  return c;
}

// ---------------------------------------------------------------------------
// One-shot weight convert: fp32 [K,N] -> B-swizzled bf16 hi/lo.
// ---------------------------------------------------------------------------
__global__ __launch_bounds__(256) void conv_w_kernel(const float* __restrict__ W,
                                                     __bf16* __restrict__ hi,
                                                     __bf16* __restrict__ lo,
                                                     int Kd, int Nn) {
  const int n = blockIdx.x * 256 + threadIdx.x;
  const int k = blockIdx.y;
  const float w = W[(size_t)k * Nn + n];
  __bf16 h, l;
  split_bf16(w, h, l);
  const size_t a = b_sw(k, n, Kd);
  hi[a] = h;
  lo[a] = l;
}

// ---------------------------------------------------------------------------
// LayerNorm (ddof=1): fp32 row in -> A-swizzled bf16 hi/lo out.
// ---------------------------------------------------------------------------
__global__ __launch_bounds__(256) void ln_kernel(const float* __restrict__ x,
                                                 const float* __restrict__ gam,
                                                 const float* __restrict__ sft,
                                                 __bf16* __restrict__ ohi,
                                                 __bf16* __restrict__ olo) {
  const int row = blockIdx.x;
  const int t = threadIdx.x;
  const float4 v = ((const float4*)(x + (size_t)row * Dm))[t];
  float sum = v.x + v.y + v.z + v.w;
  float sq = v.x * v.x + v.y * v.y + v.z * v.z + v.w * v.w;
#pragma unroll
  for (int o = 16; o > 0; o >>= 1) {
    sum += __shfl_xor(sum, o, 32);
    sq += __shfl_xor(sq, o, 32);
  }
  __shared__ float rs[8], rq[8];
  if ((t & 31) == 0) { rs[t >> 5] = sum; rq[t >> 5] = sq; }
  __syncthreads();
  sum = 0.f; sq = 0.f;
#pragma unroll
  for (int i = 0; i < 8; ++i) { sum += rs[i]; sq += rq[i]; }
  const float mean = sum * (1.0f / Dm);
  const float var = (sq - (float)Dm * mean * mean) * (1.0f / (Dm - 1));
  const float inv = rsqrtf(var + 1e-12f);
  const float4 gg = ((const float4*)gam)[t];
  const float4 ss = ((const float4*)sft)[t];
  float o4[4];
  o4[0] = gg.x * (v.x - mean) * inv + ss.x;
  o4[1] = gg.y * (v.y - mean) * inv + ss.y;
  o4[2] = gg.z * (v.z - mean) * inv + ss.z;
  o4[3] = gg.w * (v.w - mean) * inv + ss.w;
  // 4 consecutive k share strip+lane; e is consecutive -> one 8B store/limb.
  const size_t base = a_sw(row, t * 4, Dm);
  union { __bf16 b[4]; uint2 u; } ph, pl;
#pragma unroll
  for (int j = 0; j < 4; ++j) split_bf16(o4[j], ph.b[j], pl.b[j]);
  *(uint2*)&ohi[base] = ph.u;
  *(uint2*)&olo[base] = pl.u;
}

// ---------------------------------------------------------------------------
// GEMM on pre-swizzled bf16: C[M,N] = act(A @ W + bias) (+res).
// Block tile 128x64, BK=32, 512 threads = 16 waves; wave (wm=w&7, wn=w>>3)
// owns a 16x32 output strip (acc[2], 6 WMMAs per k-step).
// Ping-pong LDS double buffering, one barrier per k-step; each thread stages
// only 3 float4 (2 A chunks contiguous + 1 W chunk) so nothing spills.
// Epilogue: optional fp32 out, optional swizzled bf16 out
// (swmode 1 = A-swizzle, 2 = K-swizzle, 3 = V-swizzle).
// ---------------------------------------------------------------------------
__global__ __launch_bounds__(512) void gemm_sw_kernel(
    const __bf16* __restrict__ Ahi, const __bf16* __restrict__ Alo,
    const __bf16* __restrict__ Whi, const __bf16* __restrict__ Wlo,
    const float* __restrict__ bias, const float* __restrict__ res,
    float* __restrict__ Cf, __bf16* __restrict__ Chi, __bf16* __restrict__ Clo,
    int Nn, int Kd, int act, int swmode) {
  __shared__ alignas(32) __bf16 Asw[2 * 8192];  // [buf][limb][rowstrip0..7][512]
  __shared__ alignas(32) __bf16 Bsw[2 * 4096];  // [buf][limb][nstrip0..3][512]
  const int bm = blockIdx.y * 128;
  const int bn = blockIdx.x * 64;
  const int t = threadIdx.x;
  const int wave = t >> 5, lane = t & 31;
  const int wm = wave & 7, wn = wave >> 3;
  const int Ks = Kd >> 5;

  // Per-thread staging coordinates. A: thread t owns chunks 2t, 2t+1 which
  // are 32 contiguous bytes of one strip (2t even => no strip/limb crossing).
  const __bf16* agp;  int alds;
  {
    const int f = t * 2, limb = f >> 9, f5 = f & 511;
    const int s = f5 >> 6, off = (f5 & 63) << 3;
    agp = (limb ? Alo : Ahi) + (size_t)((bm >> 4) + s) * Ks * 512 + off;
    alds = limb * 4096 + s * 512 + off;
  }
  const __bf16* wgp;  int wlds;
  {
    const int limb = t >> 8, f4 = t & 255;
    const int s = f4 >> 6, off = (f4 & 63) << 3;
    wgp = (limb ? Wlo : Whi) + (size_t)((bn >> 4) + s) * Ks * 512 + off;
    wlds = limb * 2048 + s * 512 + off;
  }

  // Prologue: stage tile 0 into buffer 0.
  float4 ra0 = *(const float4*)agp;
  float4 ra1 = *(const float4*)(agp + 8);
  float4 rw0 = *(const float4*)wgp;
  *(float4*)&Asw[alds] = ra0;
  *(float4*)&Asw[alds + 8] = ra1;
  *(float4*)&Bsw[wlds] = rw0;

  v8f acc0 = {}, acc1 = {};
  for (int kt = 0; kt < Ks; ++kt) {
    const int p = kt & 1;
    __syncthreads();  // buffer p fully written; all reads of buffer p^1 done
    if (kt + 1 < Ks) {  // issue next tile's loads; they fly during the WMMAs
      const size_t adv = (size_t)(kt + 1) * 512;
      ra0 = *(const float4*)(agp + adv);
      ra1 = *(const float4*)(agp + adv + 8);
      rw0 = *(const float4*)(wgp + adv);
    }
    const int ab = p * 8192, wb = p * 4096;
    const v16bf ah = *(const v16bf*)&Asw[ab + wm * 512 + lane * 16];
    const v16bf al = *(const v16bf*)&Asw[ab + 4096 + wm * 512 + lane * 16];
    {
      const int st = wn * 2;
      const v16bf bh = *(const v16bf*)&Bsw[wb + st * 512 + lane * 16];
      const v16bf bl = *(const v16bf*)&Bsw[wb + 2048 + st * 512 + lane * 16];
      acc0 = wmma3(ah, al, bh, bl, acc0);
    }
    {
      const int st = wn * 2 + 1;
      const v16bf bh = *(const v16bf*)&Bsw[wb + st * 512 + lane * 16];
      const v16bf bl = *(const v16bf*)&Bsw[wb + 2048 + st * 512 + lane * 16];
      acc1 = wmma3(ah, al, bh, bl, acc1);
    }
    if (kt + 1 < Ks) {  // drain into the other buffer (nobody reads it now)
      const int qb_ = p ^ 1;
      *(float4*)&Asw[qb_ * 8192 + alds] = ra0;
      *(float4*)&Asw[qb_ * 8192 + alds + 8] = ra1;
      *(float4*)&Bsw[qb_ * 4096 + wlds] = rw0;
    }
  }

  // Epilogue. C layout: VGPR r -> row r (lanes 0-15) / r+8 (lanes 16-31).
  const int rbase = bm + wm * 16 + ((lane >> 4) << 3);
  const int ncol = bn + wn * 32 + (lane & 15);
#pragma unroll
  for (int r = 0; r < 8; ++r) {
    const int row = rbase + r;
#pragma unroll
    for (int nt = 0; nt < 2; ++nt) {
      const int col = ncol + nt * 16;
      float val = (nt == 0 ? acc0[r] : acc1[r]) + bias[col];
      if (act == 1) {
        const float u = val;
        val = 0.5f * u * (1.0f + tanhf(0.7978845608028654f * (u + 0.044715f * u * u * u)));
      }
      if (res) val += res[(size_t)row * Nn + col];
      if (Cf) Cf[(size_t)row * Nn + col] = val;
      if (Chi) {
        size_t ad;
        if (swmode == 1)      ad = a_sw(row, col, Nn);
        else if (swmode == 2) ad = k_sw(row, col);
        else                  ad = v_sw(row, col);
        __bf16 h, l;
        split_bf16(val, h, l);
        Chi[ad] = h;
        Clo[ad] = l;
      }
    }
  }
}

// ---------------------------------------------------------------------------
// Causal flash attention. One workgroup (4 waves) per (b, h, 64-row q block).
// Q/K/V fragments come straight from pre-swizzled global bf16 (32B loads).
// P lives in a per-wave LDS strip: written and read only by the same wave,
// and same-wave LDS ops complete in order (ISA 7.3), so no barriers needed.
// ---------------------------------------------------------------------------
__global__ __launch_bounds__(128) void attn_kernel(
    const __bf16* __restrict__ qhi, const __bf16* __restrict__ qlo,
    const __bf16* __restrict__ khi, const __bf16* __restrict__ klo,
    const __bf16* __restrict__ vhi, const __bf16* __restrict__ vlo,
    __bf16* __restrict__ chi, __bf16* __restrict__ clo) {
  __shared__ alignas(32) __bf16 Ph[4096];  // 64x64 P tile, A-swizzled
  __shared__ alignas(32) __bf16 Pl[4096];
  const int qblk = blockIdx.x, h = blockIdx.y, b = blockIdx.z;
  const int t = threadIdx.x, wave = t >> 5, lane = t & 31;
  const int seqstrip = b * (Ss >> 4);  // row-strip base of this batch

  // Persistent Q fragments (A operand), 2 hd k-steps.
  v16bf qah[2], qal[2];
#pragma unroll
  for (int ks = 0; ks < 2; ++ks) {
    const size_t si =
        ((size_t)(seqstrip + qblk * 4 + wave) * (Dm >> 5) + h * 2 + ks) * 512 + lane * 16;
    qah[ks] = *(const v16bf*)&qhi[si];
    qal[ks] = *(const v16bf*)&qlo[si];
  }

  float mrow[8], lrow[8];
#pragma unroll
  for (int r = 0; r < 8; ++r) { mrow[r] = -3.0e38f; lrow[r] = 0.0f; }
  v8f cacc[4] = {};
  const int rhalf = ((lane >> 4) << 3);

  for (int jb = 0; jb <= qblk; ++jb) {
    // Scores S = Q @ K^T: K fragments direct from swizzled global.
    v8f sacc[4];
#pragma unroll
    for (int nt = 0; nt < 4; ++nt) {
      v8f a = {};
#pragma unroll
      for (int ks = 0; ks < 2; ++ks) {
        const size_t si =
            ((size_t)(seqstrip + jb * 4 + nt) * (Dm >> 5) + h * 2 + ks) * 512 + lane * 16;
        const v16bf bh = *(const v16bf*)&khi[si];
        const v16bf bl = *(const v16bf*)&klo[si];
        a = wmma3(qah[ks], qal[ks], bh, bl, a);
      }
      sacc[nt] = a;
    }

    // Scale, causal mask (diagonal block only), online softmax.
    const bool diag = (jb == qblk);
#pragma unroll
    for (int r = 0; r < 8; ++r) {
      const int qrow = qblk * 64 + wave * 16 + rhalf + r;
      float mnew = mrow[r];
#pragma unroll
      for (int nt = 0; nt < 4; ++nt) {
        float s = sacc[nt][r] * 0.125f;  // 1/sqrt(64)
        if (diag) {
          const int key = jb * 64 + nt * 16 + (lane & 15);
          if (key > qrow) s = -3.0e38f;
        }
        sacc[nt][r] = s;
        mnew = fmaxf(mnew, s);
      }
#pragma unroll
      for (int o = 1; o < 16; o <<= 1) mnew = fmaxf(mnew, __shfl_xor(mnew, o, 32));
      const float alpha = __expf(mrow[r] - mnew);
      float psum = 0.f;
#pragma unroll
      for (int nt = 0; nt < 4; ++nt) {
        const float p = __expf(sacc[nt][r] - mnew);
        sacc[nt][r] = p;
        psum += p;
      }
#pragma unroll
      for (int o = 1; o < 16; o <<= 1) psum += __shfl_xor(psum, o, 32);
      lrow[r] = lrow[r] * alpha + psum;
      mrow[r] = mnew;
#pragma unroll
      for (int nt = 0; nt < 4; ++nt) cacc[nt][r] *= alpha;
    }

    // Split P once per element into this wave's A-swizzled LDS strip.
#pragma unroll
    for (int r = 0; r < 8; ++r) {
      const int prow = wave * 16 + rhalf + r;
#pragma unroll
      for (int nt = 0; nt < 4; ++nt) {
        const int key = nt * 16 + (lane & 15);
        const int kk = key & 31;
        const int ad = ((prow >> 4) * 2 + (key >> 5)) * 512 +
                       ((prow & 15) + ((kk >> 4) << 4)) * 16 +
                       (kk & 15);
        __bf16 hb, lb;
        split_bf16(sacc[nt][r], hb, lb);
        Ph[ad] = hb;
        Pl[ad] = lb;
      }
    }

    // ctx += P @ V: V fragments direct from swizzled global.
#pragma unroll
    for (int ks = 0; ks < 2; ++ks) {
      const v16bf pah = *(const v16bf*)&Ph[(wave * 2 + ks) * 512 + lane * 16];
      const v16bf pal = *(const v16bf*)&Pl[(wave * 2 + ks) * 512 + lane * 16];
#pragma unroll
      for (int nt = 0; nt < 4; ++nt) {
        const size_t si =
            (((size_t)(b * Hh + h) * 4 + nt) * (Ss >> 5) + jb * 2 + ks) * 512 + lane * 16;
        const v16bf bh = *(const v16bf*)&vhi[si];
        const v16bf bl = *(const v16bf*)&vlo[si];
        cacc[nt] = wmma3(pah, pal, bh, bl, cacc[nt]);
      }
    }
  }

  // Normalize; write ctx A-swizzled (ready as O-GEMM A operand).
#pragma unroll
  for (int r = 0; r < 8; ++r) {
    const int row = b * Ss + qblk * 64 + wave * 16 + rhalf + r;
    const float invl = 1.0f / lrow[r];
#pragma unroll
    for (int nt = 0; nt < 4; ++nt) {
      const int col = h * 64 + nt * 16 + (lane & 15);
      const size_t ad = a_sw(row, col, Dm);
      __bf16 hb, lb;
      split_bf16(cacc[nt][r] * invl, hb, lb);
      chi[ad] = hb;
      clo[ad] = lb;
    }
  }
}

// ---------------------------------------------------------------------------
extern "C" void kernel_launch(void* const* d_in, const int* in_sizes, int n_in,
                              void* d_out, int out_size, void* d_ws, size_t ws_size,
                              hipStream_t stream) {
  (void)in_sizes; (void)n_in; (void)out_size; (void)ws_size;
  const float* x  = (const float*)d_in[0];
  const float* Wq = (const float*)d_in[1];
  const float* bq = (const float*)d_in[2];
  const float* Wk = (const float*)d_in[3];
  const float* bk = (const float*)d_in[4];
  const float* Wv = (const float*)d_in[5];
  const float* bv = (const float*)d_in[6];
  const float* Wo = (const float*)d_in[7];
  const float* bo = (const float*)d_in[8];
  const float* W1 = (const float*)d_in[9];
  const float* b1 = (const float*)d_in[10];
  const float* W2 = (const float*)d_in[11];
  const float* b2 = (const float*)d_in[12];
  const float* g1 = (const float*)d_in[13];
  const float* s1 = (const float*)d_in[14];
  const float* g2 = (const float*)d_in[15];
  const float* s2 = (const float*)d_in[16];
  float* out = (float*)d_out;

  char* p = (char*)d_ws;
  auto carve = [&](size_t nelem) -> __bf16* {
    __bf16* r = (__bf16*)p;
    p += (nelem * 2 + 255) & ~(size_t)255;
    return r;
  };
  const size_t TA = (size_t)Mrows * Dm;
  __bf16 *h1h = carve(TA), *h1l = carve(TA);
  __bf16 *qh = carve(TA), *ql = carve(TA);
  __bf16 *kh = carve(TA), *kl = carve(TA);
  __bf16 *vh = carve(TA), *vl = carve(TA);
  __bf16 *ch = carve(TA), *cl = carve(TA);
  __bf16 *h2h = carve(TA), *h2l = carve(TA);
  __bf16 *ffh = carve((size_t)Mrows * DFFn), *ffl = carve((size_t)Mrows * DFFn);
  __bf16 *wqh = carve((size_t)Dm * Dm), *wql = carve((size_t)Dm * Dm);
  __bf16 *wkh = carve((size_t)Dm * Dm), *wkl = carve((size_t)Dm * Dm);
  __bf16 *wvh = carve((size_t)Dm * Dm), *wvl = carve((size_t)Dm * Dm);
  __bf16 *woh = carve((size_t)Dm * Dm), *wol = carve((size_t)Dm * Dm);
  __bf16 *w1h = carve((size_t)Dm * DFFn), *w1l = carve((size_t)Dm * DFFn);
  __bf16 *w2h = carve((size_t)DFFn * Dm), *w2l = carve((size_t)DFFn * Dm);
  float* x1 = (float*)p;
  p += TA * 4;

  const dim3 blk(256);
  const dim3 blkG(512);
  // One-shot weight convert + swizzle.
  conv_w_kernel<<<dim3(Dm / 256, Dm), blk, 0, stream>>>(Wq, wqh, wql, Dm, Dm);
  conv_w_kernel<<<dim3(Dm / 256, Dm), blk, 0, stream>>>(Wk, wkh, wkl, Dm, Dm);
  conv_w_kernel<<<dim3(Dm / 256, Dm), blk, 0, stream>>>(Wv, wvh, wvl, Dm, Dm);
  conv_w_kernel<<<dim3(Dm / 256, Dm), blk, 0, stream>>>(Wo, woh, wol, Dm, Dm);
  conv_w_kernel<<<dim3(DFFn / 256, Dm), blk, 0, stream>>>(W1, w1h, w1l, Dm, DFFn);
  conv_w_kernel<<<dim3(Dm / 256, DFFn), blk, 0, stream>>>(W2, w2h, w2l, DFFn, Dm);

  const dim3 gD(Dm / 64, Mrows / 128);
  const dim3 gF(DFFn / 64, Mrows / 128);

  // 1. pre-norm -> A-swizzled bf16
  ln_kernel<<<Mrows, blk, 0, stream>>>(x, g1, s1, h1h, h1l);
  // 2-4. QKV projections, outputs pre-swizzled for attention
  gemm_sw_kernel<<<gD, blkG, 0, stream>>>(h1h, h1l, wqh, wql, bq, nullptr,
                                          nullptr, qh, ql, Dm, Dm, 0, 1);
  gemm_sw_kernel<<<gD, blkG, 0, stream>>>(h1h, h1l, wkh, wkl, bk, nullptr,
                                          nullptr, kh, kl, Dm, Dm, 0, 2);
  gemm_sw_kernel<<<gD, blkG, 0, stream>>>(h1h, h1l, wvh, wvl, bv, nullptr,
                                          nullptr, vh, vl, Dm, Dm, 0, 3);
  // 5. causal flash attention -> A-swizzled ctx
  attn_kernel<<<dim3(Ss / 64, Hh, Bb), dim3(128), 0, stream>>>(qh, ql, kh, kl,
                                                               vh, vl, ch, cl);
  // 6. output projection + residual (fp32 x1 = x + ctx @ Wo + bo)
  gemm_sw_kernel<<<gD, blkG, 0, stream>>>(ch, cl, woh, wol, bo, x,
                                          x1, nullptr, nullptr, Dm, Dm, 0, 0);
  // 7. second norm -> A-swizzled bf16
  ln_kernel<<<Mrows, blk, 0, stream>>>(x1, g2, s2, h2h, h2l);
  // 8. MLP up + GELU -> A-swizzled ff
  gemm_sw_kernel<<<gF, blkG, 0, stream>>>(h2h, h2l, w1h, w1l, b1, nullptr,
                                          nullptr, ffh, ffl, DFFn, Dm, 1, 1);
  // 9. MLP down + residual -> final fp32 output
  gemm_sw_kernel<<<gD, blkG, 0, stream>>>(ffh, ffl, w2h, w2l, b2, x1,
                                          out, nullptr, nullptr, Dm, DFFn, 0, 0);
}
